// ScalarUpdater_1924145349108
// MI455X (gfx1250) — compile-verified, matrix operand order checked by reading, not run on hardware
//
#include <hip/hip_runtime.h>
#include <hip/hip_bf16.h>

// ---------------------------------------------------------------------------
// ScalarUpdater for MI455X (gfx1250, wave32, WMMA).
//
// Structure exploited: fts depends only on (edge_state_idx, node_state_idx),
// each in [0,16). => only 256 distinct fts rows. The big (1M x 256) @ (256x128)
// GEMM collapses to a 256x256x128 GEMM done with v_wmma_f32_16x16x32_f16,
// followed by a 256x4 gate table and a memory-minimal per-edge gather/scatter.
// ---------------------------------------------------------------------------

typedef __attribute__((ext_vector_type(16))) _Float16 v16h;
typedef __attribute__((ext_vector_type(8)))  float    v8f;

#define HDIM 128

// ---------- kernel 1: zero accumulators -------------------------------------
__global__ void zero_kernel(float* __restrict__ acc, float* __restrict__ loss_accum, int N) {
    int i = blockIdx.x * blockDim.x + threadIdx.x;
    if (i < N) acc[i] = 0.0f;
    if (i == 0) *loss_accum = 0.0f;
}

// ---------- kernel 2: 4-bit binary-state indices ----------------------------
__global__ void state_idx_kernel(const int* __restrict__ node_states,
                                 const int* __restrict__ edge_states,
                                 int* __restrict__ nsidx, int* __restrict__ esidx,
                                 int N, int E) {
    int i = blockIdx.x * blockDim.x + threadIdx.x;
    if (i < E) {
        const int* r = edge_states + i * 4;
        esidx[i] = r[0] + 2 * r[1] + 4 * r[2] + 8 * r[3];
    }
    if (i < N) {
        const int* r = node_states + i * 4;
        nsidx[i] = r[0] + 2 * r[1] + 4 * r[2] + 8 * r[3];
    }
}

// ---------- kernel 3: table[256][128] = concat(edge_emb,node_emb) @ W + b ----
// One wave (32 threads) per 16x16 output tile; M=256 combos, N=128, K=256.
// A/B fragments assembled per CDNA5 16-bit WMMA layouts:
//   lanes 0-15 : K = i&7 (+16 if i>=8)         (v16h element i)
//   lanes 16-31: same +8
// C/D f32 layout: VGPR r -> row M = r + 8*(lane>=16), col N = lane&15.
__global__ __launch_bounds__(32)
void table_wmma_kernel(const float* __restrict__ edge_emb,
                       const float* __restrict__ node_emb,
                       const float* __restrict__ combine_W,   // (256,128) row-major
                       const float* __restrict__ combine_b,   // (128)
                       float* __restrict__ table) {           // (256,128)
    const int nt   = blockIdx.x & 7;    // 8 column tiles
    const int mt   = blockIdx.x >> 3;   // 16 row tiles
    const int lane = threadIdx.x;
    const int half = lane >> 4;
    const int l16  = lane & 15;
    const int m    = mt * 16 + l16;     // combo row for A fragment
    const int eidx = m >> 4;
    const int nidx = m & 15;
    const int n    = nt * 16 + l16;     // output column for B fragment

    v8f acc = {};
    for (int k0 = 0; k0 < 2 * HDIM; k0 += 32) {
        v16h a, b;
#pragma unroll
        for (int i = 0; i < 16; ++i) {
            const int k = k0 + (i & 7) + ((i >> 3) << 4) + (half << 3);
            const float av = (k < HDIM) ? edge_emb[eidx * HDIM + k]
                                        : node_emb[nidx * HDIM + (k - HDIM)];
            a[i] = (_Float16)av;
            b[i] = (_Float16)combine_W[k * HDIM + n];
        }
        acc = __builtin_amdgcn_wmma_f32_16x16x32_f16(
            /*neg_a=*/false, a, /*neg_b=*/false, b,
            /*c_mod=*/(short)0, acc, /*reuse_a=*/false, /*reuse_b=*/false);
    }
    const float bias = combine_b[n];
#pragma unroll
    for (int r = 0; r < 8; ++r) {
        const int row = mt * 16 + r + half * 8;
        table[row * HDIM + n] = acc[r] + bias;
    }
}

// ---------- kernel 4: gate table gates[256][4] -------------------------------
// pair_gumbel_first(W,b) = softmax((h@W+b)/tau)[:,0] = 1/(1+exp((l1-l0)/tau))
__global__ __launch_bounds__(256)
void gates_kernel(const float* __restrict__ table,
                  const float* __restrict__ incW, const float* __restrict__ incB,
                  const float* __restrict__ keepW, const float* __restrict__ keepB,
                  const float* __restrict__ pushW, const float* __restrict__ pushB,
                  const float* __restrict__ pushNW, const float* __restrict__ pushNB,
                  const int* __restrict__ training_step,
                  float* __restrict__ gates) {
    const int c = blockIdx.x * blockDim.x + threadIdx.x;
    if (c >= 256) return;

    // temp_by_step(training_step, 1.0, 0.1, 10000, 0.1)
    float tau;
    const int ts = training_step[0];
    if (ts == -1) {
        tau = 0.1f;
    } else {
        float frac = fminf((float)ts / 10000.0f, 1.0f);
        tau = 1.0f + (0.1f - 1.0f) * frac;
    }
    const float inv_tau = 1.0f / tau;

    const float* h = table + c * HDIM;
    float d0[4] = {0.f, 0.f, 0.f, 0.f};
    float d1[4] = {0.f, 0.f, 0.f, 0.f};
    for (int i = 0; i < HDIM; ++i) {
        const float hv = h[i];
        d0[0] += hv * incW[i * 2 + 0];  d1[0] += hv * incW[i * 2 + 1];
        d0[1] += hv * keepW[i * 2 + 0]; d1[1] += hv * keepW[i * 2 + 1];
        d0[2] += hv * pushW[i * 2 + 0]; d1[2] += hv * pushW[i * 2 + 1];
        d0[3] += hv * pushNW[i * 2 + 0];d1[3] += hv * pushNW[i * 2 + 1];
    }
    const float b0[4] = {incB[0], keepB[0], pushB[0], pushNB[0]};
    const float b1[4] = {incB[1], keepB[1], pushB[1], pushNB[1]};
#pragma unroll
    for (int hd = 0; hd < 4; ++hd) {
        const float dl = ((d1[hd] + b1[hd]) - (d0[hd] + b0[hd])) * inv_tau;
        gates[c * 4 + hd] = 1.0f / (1.0f + __expf(dl));
    }
}

// ---------- kernel 5: per-edge gather / gate lookup / scatter ---------------
__global__ __launch_bounds__(256)
void edge_kernel(const float* __restrict__ scalars,
                 const int* __restrict__ edge_index,   // [0,E)=src, [E,2E)=dst
                 const int* __restrict__ brev,
                 const int* __restrict__ esidx,
                 const int* __restrict__ nsidx,
                 const float* __restrict__ gates,      // 256 x 4
                 float* __restrict__ out,
                 float* __restrict__ acc,
                 int N, int E) {
    const int e = blockIdx.x * blockDim.x + threadIdx.x;
    if (e >= E) return;
    const int src = edge_index[e];
    const int dst = edge_index[E + e];
    const int c   = esidx[brev[e]] * 16 + nsidx[src];
    const float4 g = reinterpret_cast<const float4*>(gates)[c]; // inc,keep,push_wo,push_w
    const float se   = scalars[e];
    const float s_wo = se - scalars[dst];     // dst < N, node_scalars = scalars[:N]
    const float s_w  = s_wo + scalars[src];
    atomicAdd(&acc[dst], g.z * s_wo + g.w * s_w);
    out[e] = g.x + se * g.y;                  // increment + keep (edge_push added later)
}

// ---------- kernel 6: add edge_push, block-reduced MSE ----------------------
__global__ __launch_bounds__(256)
void finalize_kernel(float* __restrict__ out,
                     const float* __restrict__ acc,
                     const float* __restrict__ target,  // batch_scalars[:,0,0]
                     float* __restrict__ loss_accum,
                     int N, int E) {
    __shared__ float red[256];
    const int e = blockIdx.x * blockDim.x + threadIdx.x;
    float d2 = 0.0f;
    if (e < E) {
        float v = out[e];
        if (e < N) { v += acc[e]; out[e] = v; }
        const float d = target[e] - v;
        d2 = d * d;
    }
    red[threadIdx.x] = d2;
    __syncthreads();
    for (int s = 128; s > 0; s >>= 1) {
        if (threadIdx.x < s) red[threadIdx.x] += red[threadIdx.x + s];
        __syncthreads();
    }
    if (threadIdx.x == 0) atomicAdd(loss_accum, red[0]);
}

// ---------- kernel 7: write loss -------------------------------------------
__global__ void loss_kernel(float* __restrict__ out, const float* __restrict__ loss_accum, int E) {
    out[E] = *loss_accum / (float)E;
}

// ---------------------------------------------------------------------------
extern "C" void kernel_launch(void* const* d_in, const int* in_sizes, int n_in,
                              void* d_out, int out_size, void* d_ws, size_t ws_size,
                              hipStream_t stream) {
    // Inputs (setup_inputs dict order)
    const int*   node_states   = (const int*)  d_in[0];   // N*4
    const int*   edge_states   = (const int*)  d_in[1];   // E*4
    const float* scalars       = (const float*)d_in[2];   // E
    const int*   edge_index    = (const int*)  d_in[3];   // 2*E
    const int*   brev          = (const int*)  d_in[4];   // E
    const float* batch_scalars = (const float*)d_in[5];   // E (x1x1)
    // d_in[6] = processor_step (always indexes the size-1 axis -> unused)
    const int*   training_step = (const int*)  d_in[7];
    const float* node_emb      = (const float*)d_in[8];   // 16*128
    const float* edge_emb      = (const float*)d_in[9];   // 16*128
    const float* combine_W     = (const float*)d_in[10];  // 256*128
    const float* combine_b     = (const float*)d_in[11];  // 128
    const float* keep_W        = (const float*)d_in[12];  // 128*2
    const float* keep_b        = (const float*)d_in[13];
    const float* push_W        = (const float*)d_in[14];
    const float* push_b        = (const float*)d_in[15];
    const float* push_node_W   = (const float*)d_in[16];
    const float* push_node_b   = (const float*)d_in[17];
    const float* increment_W   = (const float*)d_in[18];
    const float* increment_b   = (const float*)d_in[19];

    const int N = in_sizes[0] / 4;
    const int E = in_sizes[2];
    float* out = (float*)d_out;   // E scalars + 1 loss

    // Workspace carve-up (256B-aligned regions)
    char*  wsb = (char*)d_ws;
    size_t off = 0;
    auto take = [&](size_t bytes) -> void* {
        void* p = wsb + off;
        off += (bytes + 255) & ~(size_t)255;
        return p;
    };
    int*   esidx      = (int*)  take(sizeof(int) * (size_t)E);
    int*   nsidx      = (int*)  take(sizeof(int) * (size_t)N);
    float* table      = (float*)take(sizeof(float) * 256 * HDIM);
    float* gates      = (float*)take(sizeof(float) * 256 * 4);
    float* acc        = (float*)take(sizeof(float) * (size_t)N);
    float* loss_accum = (float*)take(sizeof(float));
    (void)ws_size; (void)n_in; (void)out_size;

    const int TPB = 256;
    const int gridE = (E + TPB - 1) / TPB;
    const int gridN = (N + TPB - 1) / TPB;

    zero_kernel<<<gridN, TPB, 0, stream>>>(acc, loss_accum, N);
    state_idx_kernel<<<gridE, TPB, 0, stream>>>(node_states, edge_states, nsidx, esidx, N, E);
    table_wmma_kernel<<<128, 32, 0, stream>>>(edge_emb, node_emb, combine_W, combine_b, table);
    gates_kernel<<<1, 256, 0, stream>>>(table,
                                        increment_W, increment_b,
                                        keep_W, keep_b,
                                        push_W, push_b,
                                        push_node_W, push_node_b,
                                        training_step, gates);
    edge_kernel<<<gridE, TPB, 0, stream>>>(scalars, edge_index, brev, esidx, nsidx,
                                           gates, out, acc, N, E);
    finalize_kernel<<<gridE, TPB, 0, stream>>>(out, acc, batch_scalars, loss_accum, N, E);
    loss_kernel<<<1, 1, 0, stream>>>(out, loss_accum, E);
}
